// A3TGCN_2_73512660238941
// MI455X (gfx1250) — compile-verified, matrix operand order checked by reading, not run on hardware
//
#include <hip/hip_runtime.h>

typedef __attribute__((ext_vector_type(16))) _Float16 v16h;
typedef __attribute__((ext_vector_type(8)))  float    v8f;

#define Bb  16
#define Nn  10000
#define Fin 2
#define Tt  12
#define Ee  160000
#define Fo  32

// workspace layout (float offsets)
#define OFF_COEF 0              // 256 floats: Mz0,Mz1,cz,Mh0,Mh1,ch (6x32) + probs(12)
#define OFF_DEG  256            // 10000
#define OFF_DINV 10496          // 10000
#define OFF_XT   20992          // N*B*24 = 3,840,000
#define OFF_AGG  3861504        // N*B*24 = 3,840,000
// total ~7.70M floats ~= 30.8 MB — fits easily in L2 (192 MB)

__device__ __forceinline__ void atomAddF(float* p, float v) {
  unsafeAtomicAdd(p, v);  // lowers to global_atomic_add_f32
}

__device__ __forceinline__ float fastRcp(float x) {
  return __builtin_amdgcn_rcpf(x);  // v_rcp_f32
}

// Fold GRU weights: Mz = W_z @ lz_W[:32], cz = b_z @ lz_W[:32] + lz_b (same for h-gate);
// also softmax of attention logits.
__global__ void k_pre(const float* __restrict__ Wz, const float* __restrict__ bz,
                      const float* __restrict__ Wh, const float* __restrict__ bh,
                      const float* __restrict__ lzW, const float* __restrict__ lzb,
                      const float* __restrict__ lhW, const float* __restrict__ lhb,
                      const float* __restrict__ att, float* __restrict__ coef) {
  int t = threadIdx.x;
  if (t < 32) {
    float mz0 = 0.f, mz1 = 0.f, cz = 0.f, mh0 = 0.f, mh1 = 0.f, ch = 0.f;
    for (int k = 0; k < 32; ++k) {
      float lz = lzW[k * 32 + t];
      float lh = lhW[k * 32 + t];
      mz0 += Wz[k]      * lz;
      mz1 += Wz[32 + k] * lz;
      cz  += bz[k]      * lz;
      mh0 += Wh[k]      * lh;
      mh1 += Wh[32 + k] * lh;
      ch  += bh[k]      * lh;
    }
    coef[t]       = mz0;
    coef[32 + t]  = mz1;
    coef[64 + t]  = cz + lzb[t];
    coef[96 + t]  = mh0;
    coef[128 + t] = mh1;
    coef[160 + t] = ch + lhb[t];
  } else if (t == 32) {
    float m = att[0];
    for (int i = 1; i < Tt; ++i) m = fmaxf(m, att[i]);
    float e[Tt]; float s = 0.f;
    for (int i = 0; i < Tt; ++i) { e[i] = __expf(att[i] - m); s += e[i]; }
    float inv = 1.f / s;
    for (int i = 0; i < Tt; ++i) coef[192 + i] = e[i] * inv;
  }
}

__global__ void k_deg_init(float* __restrict__ deg) {
  int n = blockIdx.x * blockDim.x + threadIdx.x;
  if (n < Nn) deg[n] = 1.0f;  // self loop
}

__global__ void k_deg_edges(const int* __restrict__ ei, float* __restrict__ deg) {
  int e = blockIdx.x * blockDim.x + threadIdx.x;
  if (e < Ee) atomAddF(&deg[ei[Ee + e]], 1.0f);
}

__global__ void k_dinv(const float* __restrict__ deg, float* __restrict__ dinv) {
  int n = blockIdx.x * blockDim.x + threadIdx.x;
  if (n < Nn) dinv[n] = __frsqrt_rn(deg[n]);
}

// x [B,N,F,T] -> node-major xT [N, B*F*T=384]; also seed agg with the
// self-loop contribution dinv[n]^2 * x.
__global__ void k_transpose(const float* __restrict__ x, const float* __restrict__ dinv,
                            float* __restrict__ xT, float* __restrict__ agg) {
  int i = blockIdx.x * blockDim.x + threadIdx.x;
  if (i >= Bb * Nn * Fin * Tt) return;
  int b   = i / (Nn * 24);
  int rem = i - b * (Nn * 24);
  int n   = rem / 24;
  int c   = rem - n * 24;
  float v  = x[i];
  float di = dinv[n];
  size_t o = (size_t)n * 384 + b * 24 + c;
  xT[o]  = v;
  agg[o] = di * di * v;
}

// One wave per edge: gather src row (384 contiguous floats), scatter-add into
// dst row with symmetric GCN norm. All traffic is L2-resident (~0.5 GB total),
// x/agg (31 MB) stay resident in the 192 MB L2.
__global__ void k_scatter(const int* __restrict__ ei, const float* __restrict__ xT,
                          const float* __restrict__ dinv, float* __restrict__ agg) {
  int gid  = blockIdx.x * blockDim.x + threadIdx.x;
  int e    = gid >> 5;
  int lane = gid & 31;
  if (e >= Ee) return;
  int s = ei[e];
  int d = ei[Ee + e];
  float w = dinv[s] * dinv[d];
  const float* __restrict__ sp = xT  + (size_t)s * 384;
  float*       __restrict__ dp = agg + (size_t)d * 384;
#pragma unroll
  for (int i = 0; i < 12; ++i) {
    int j = lane + i * 32;
    atomAddF(&dp[j], w * sp[j]);
  }
}

// 4 waves / block; one wave handles one node n == one 16-row tile (rows = batch b).
// Per lane: 16 of the 32 hidden features of one row (WMMA A-layout lane split).
// The 16-feature loop is kept ROLLED (unroll 1) and the per-feature result is
// staged as f16 into the lane's private 32B LDS slot — this keeps VGPR usage low
// (no MSB-VGPR spill zone like the fully-unrolled version) so 4+ waves/SIMD can
// hide v_exp latency. The slot is then reloaded as the WMMA A fragment and
// v_wmma_f32_16x16x32_f16 performs the final [16x32]@[32x12->16] projection.
__global__ void __launch_bounds__(128)
k_dense(const float* __restrict__ agg, const float* __restrict__ coef,
        const float* __restrict__ linW, const float* __restrict__ linb,
        float* __restrict__ out) {
  __shared__ v16h stage[128];          // 32 bytes per thread, 4 KB per block

  int lane = threadIdx.x & 31;
  int wave = threadIdx.x >> 5;
  int tile = blockIdx.x * 4 + wave;    // node index n
  int mrow = lane & 15;                // row within tile = batch index b
  int hb   = (lane >= 16) ? 8 : 0;
  int ncol = mrow;                     // output column (time period) for this lane

  // aggregated feature row for (n, b): 24 floats = [f=0..1][t=0..11]
  const float* __restrict__ up = agg + (size_t)(tile * 16 + mrow) * 24;
  float u[24];
#pragma unroll
  for (int j = 0; j < 24; ++j) u[j] = up[j];

  float probs[Tt];
#pragma unroll
  for (int t = 0; t < Tt; ++t) probs[t] = coef[192 + t];

  _Float16* st = (_Float16*)&stage[threadIdx.x];

#pragma unroll 1
  for (int i = 0; i < 16; ++i) {
    // WMMA 16-bit A/B slot->K mapping: lanes 0-15 hold K {0..7,16..23},
    // lanes 16-31 hold K {8..15,24..31}
    int k = (i < 8) ? (hb + i) : (8 + hb + i);
    float mz0 = coef[k],      mz1 = coef[32 + k],  czv = coef[64 + k];
    float mh0 = coef[96 + k], mh1 = coef[128 + k], chv = coef[160 + k];
    float acc = 0.f;
#pragma unroll
    for (int t = 0; t < Tt; ++t) {
      float u0 = u[t], u1 = u[12 + t];
      float zarg = fmaf(u1, mz1, fmaf(u0, mz0, czv));
      float harg = fmaf(u1, mh1, fmaf(u0, mh0, chv));
      // sigmoid(z) = 1/(1+exp(-z))
      float Z  = fastRcp(1.f + __expf(-zarg));
      // tanh(h) = 1 - 2/(exp(2h)+1)  (overflow-safe: exp->inf => rcp->0 => 1)
      float Ht = fmaf(-2.f, fastRcp(__expf(2.f * harg) + 1.f), 1.f);
      acc = fmaf(probs[t], (1.f - Z) * Ht, acc);  // H0==0 => H=(1-Z)*Ht
    }
    st[i] = (_Float16)fmaxf(acc, 0.f);            // relu, staged to LDS
  }

  // B fragment: lin_W [32x12] padded to 32x16, same slot->K mapping
  v16h bm;
#pragma unroll
  for (int i = 0; i < 16; ++i) {
    int k = (i < 8) ? (hb + i) : (8 + hb + i);
    bm[i] = (_Float16)((ncol < Tt) ? linW[k * Tt + ncol] : 0.f);
  }

  v16h a = stage[threadIdx.x];  // reload A fragment (same-lane LDS, in-order)

  v8f c = {};
  // D = A(16x32) * B(32x16) + 0
  c = __builtin_amdgcn_wmma_f32_16x16x32_f16(false, a, false, bm,
                                             (short)0, c, false, false);

  if (ncol < Tt) {
    float bias = linb[ncol];
    size_t obase = (size_t)tile * Tt + ncol;
#pragma unroll
    for (int r = 0; r < 8; ++r) {
      int M = r + hb;  // batch index
      out[(size_t)M * (Nn * Tt) + obase] = c[r] + bias;
    }
  }
}

extern "C" void kernel_launch(void* const* d_in, const int* in_sizes, int n_in,
                              void* d_out, int out_size, void* d_ws, size_t ws_size,
                              hipStream_t stream) {
  const float* x    = (const float*)d_in[0];
  const int*   ei   = (const int*)d_in[1];     // edge_index [2,E]
  const float* Wz   = (const float*)d_in[2];
  const float* bz   = (const float*)d_in[3];
  // d_in[4], d_in[5] (W_r, b_r) are dead: H0==0 kills the reset gate
  const float* Wh   = (const float*)d_in[6];
  const float* bh   = (const float*)d_in[7];
  const float* lzW  = (const float*)d_in[8];
  const float* lzb  = (const float*)d_in[9];
  // d_in[10], d_in[11] (lr_W, lr_b) dead for the same reason
  const float* lhW  = (const float*)d_in[12];
  const float* lhb  = (const float*)d_in[13];
  const float* att  = (const float*)d_in[14];
  const float* linW = (const float*)d_in[15];
  const float* linb = (const float*)d_in[16];

  float* ws   = (float*)d_ws;
  float* coef = ws + OFF_COEF;
  float* deg  = ws + OFF_DEG;
  float* dinv = ws + OFF_DINV;
  float* xT   = ws + OFF_XT;
  float* agg  = ws + OFF_AGG;
  float* out  = (float*)d_out;

  k_pre<<<1, 64, 0, stream>>>(Wz, bz, Wh, bh, lzW, lzb, lhW, lhb, att, coef);
  k_deg_init<<<(Nn + 255) / 256, 256, 0, stream>>>(deg);
  k_deg_edges<<<(Ee + 255) / 256, 256, 0, stream>>>(ei, deg);
  k_dinv<<<(Nn + 255) / 256, 256, 0, stream>>>(deg, dinv);
  k_transpose<<<(Bb * Nn * Fin * Tt + 255) / 256, 256, 0, stream>>>(x, dinv, xT, agg);
  k_scatter<<<(Ee * 32) / 256, 256, 0, stream>>>(ei, xT, dinv, agg);
  k_dense<<<Nn / 4, 128, 0, stream>>>(agg, coef, linW, linb, out);
}
